// NLIF_9431748182366
// MI455X (gfx1250) — matrix operand order repaired and checked by reading, not compiled, not run on hardware
//
#include <hip/hip_runtime.h>
#include <stdint.h>

// ---------------------------------------------------------------------------
// NLIF recurrent network, MI455X / gfx1250, wave32 WMMA bf16 path.
//   N=1024 neurons, B=64 batch, T=128 steps.
//   Per step: I = s@Ws + s_fast@Wf + x@W_in  (two 64x1024x1024 GEMMs)
//   Sequential over T -> one fused kernel launch per step.
// ---------------------------------------------------------------------------

typedef __attribute__((ext_vector_type(16))) __bf16 v16bf;
typedef __attribute__((ext_vector_type(8)))  float  v8f;

#define NN 1024
#define NB 64      // batch
#define NT 128     // time steps

__device__ __forceinline__ __bf16 f2bf(float f) {
    // round-to-nearest-even f32 -> bf16, pure integer ops
    uint32_t u = __builtin_bit_cast(uint32_t, f);
    u += 0x7FFFu + ((u >> 16) & 1u);
    uint16_t h = (uint16_t)(u >> 16);
    return __builtin_bit_cast(__bf16, h);
}

// ---------------------------------------------------------------------------
// One-time: mask diagonal, convert weights to bf16 packed in WMMA B-layout.
// Packed index q = ((kb*64 + nb)*32 + lane)*16 + e
//   N = nb*16 + (lane&15) ;  K = kb*32 + (lane>>4)*16 + e
// ---------------------------------------------------------------------------
__global__ void nlif_pack_weights(const float* __restrict__ Wsyn,
                                  const float* __restrict__ Wfast,
                                  __bf16* __restrict__ Wsb,
                                  __bf16* __restrict__ Wfb) {
    unsigned p = blockIdx.x * 256u + threadIdx.x;   // 0 .. 2*1024*1024-1
    unsigned which = p >> 20;
    unsigned q = p & 0xFFFFFu;
    unsigned e    = q & 15u;
    unsigned lane = (q >> 4) & 31u;
    unsigned nb   = (q >> 9) & 63u;
    unsigned kb   = q >> 15;
    unsigned n = nb * 16u + (lane & 15u);
    unsigned k = kb * 32u + ((lane >> 4) << 4) + e;
    const float* W = which ? Wfast : Wsyn;
    float v = (k == n) ? 0.0f : W[k * NN + n];      // zero self-recurrence
    (which ? Wfb : Wsb)[q] = f2bf(v);
}

// ---------------------------------------------------------------------------
// Zero initial state + readout accumulation region of d_out.
// ---------------------------------------------------------------------------
__global__ void nlif_init(float* __restrict__ s_f32,
                          __bf16* __restrict__ s_pk,
                          __bf16* __restrict__ sf_pk,
                          float* __restrict__ vpk,
                          float* __restrict__ readout) {
    int i = blockIdx.x * 256 + threadIdx.x;         // 65536 threads
    s_f32[i] = 0.0f;
    vpk[i]   = 0.0f;
    ((uint16_t*)s_pk)[i]  = 0;
    ((uint16_t*)sf_pk)[i] = 0;
    if (i < NT * NB * 2) readout[i] = 0.0f;
}

// ---------------------------------------------------------------------------
// One simulation step. 256 waves, one 16x16 (batch x neuron) tile each.
// ---------------------------------------------------------------------------
__global__ __launch_bounds__(256) void nlif_step(
    const float*  __restrict__ x_t,       // [64,2]
    const float*  __restrict__ W_in,      // [2,1024]
    const float*  __restrict__ O,         // [2,1024]
    const __bf16* __restrict__ Wsb,       // packed B-layout
    const __bf16* __restrict__ Wfb,
    const float*  __restrict__ s_prev,    // [64,1024] f32
    const __bf16* __restrict__ sPk_prev,  // packed A-layout
    const __bf16* __restrict__ sfPk_prev,
    float*  __restrict__ s_cur,
    __bf16* __restrict__ sPk_cur,
    __bf16* __restrict__ sfPk_cur,
    float*  __restrict__ vpk,             // tile-packed membrane potential
    float*  __restrict__ spikes,          // d_out + t*B*N
    float*  __restrict__ readout)         // d_out + T*B*N + t*B*2
{
    const int lane  = threadIdx.x & 31;
    const int wave  = (blockIdx.x * 256 + threadIdx.x) >> 5;  // 0..255
    const int mb    = wave >> 6;    // batch tile  0..3
    const int nb    = wave & 63;    // neuron tile 0..63
    const int khalf = lane >> 4;
    const int l15   = lane & 15;

    const v16bf* aSv = (const v16bf*)sPk_prev;
    const v16bf* aFv = (const v16bf*)sfPk_prev;
    const v16bf* bSv = (const v16bf*)Wsb;
    const v16bf* bFv = (const v16bf*)Wfb;
    const int aBase = (mb * 32) * 32 + lane;   // + kb*32 per K-block
    const int bBase = nb * 32 + lane;          // + kb*2048 per K-block

    v8f cs = {0.f, 0.f, 0.f, 0.f, 0.f, 0.f, 0.f, 0.f};
    v8f cf = {0.f, 0.f, 0.f, 0.f, 0.f, 0.f, 0.f, 0.f};

    for (int kb = 0; kb < 32; ++kb) {
        v16bf aS = aSv[aBase + kb * 32];
        v16bf aF = aFv[aBase + kb * 32];
        v16bf wS = bSv[bBase + kb * 2048];
        v16bf wF = bFv[bBase + kb * 2048];
        cs = __builtin_amdgcn_wmma_f32_16x16x32_bf16(false, aS, false, wS,
                                                     (short)0, cs, false, false);
        cf = __builtin_amdgcn_wmma_f32_16x16x32_bf16(false, aF, false, wF,
                                                     (short)0, cf, false, false);
    }

    // ---- elementwise neuron update on the C/D tile ----
    const int n = nb * 16 + l15;
    const float w0 = W_in[n], w1 = W_in[NN + n];
    const float o0 = O[n],    o1 = O[NN + n];
    float* vp = vpk + (size_t)((mb * 64 + nb) * 32 + lane) * 8;

    // A-layout coords for next step's packed state writes (K = n)
    const int kb2    = n >> 5;
    const int khalf2 = (n >> 4) & 1;
    const int e2     = n & 15;

    #pragma unroll
    for (int r = 0; r < 8; ++r) {
        const int m = mb * 16 + khalf * 8 + r;
        const float x0 = x_t[m * 2 + 0];
        const float x1 = x_t[m * 2 + 1];

        float I  = cs[r] + cf[r] + x0 * w0 + x1 * w1;   // I_TONIC = 0
        float v  = vp[r];
        float dv = (I - v) / 10.0f;                     // TAU_M
        float vn = v + dv;
        float g  = fminf(fmaxf(vn, -1.0f), 1.0f);
        float gdv = g * dv;

        float so   = s_prev[m * NN + n];
        float snew = so + (gdv - so) / 10.0f;           // TAU_S
        // s_fast_new = s_fast + (gdv - s_fast) * 1  ==  gdv

        float sp_p = (vn >=  1.0f) ? 1.0f : 0.0f;
        float sp_n = (vn <= -1.0f) ? 1.0f : 0.0f;
        float vnew = ((1.0f - sp_p) + (1.0f - sp_n)) * vn;  // faithful to ref

        vp[r] = vnew;
        s_cur[m * NN + n]      = snew;
        spikes[m * NN + n]     = sp_p + sp_n;

        const int lane2 = khalf2 * 16 + (m & 15);
        const int aidx  = ((mb * 32 + kb2) * 32 + lane2) * 16 + e2;
        sPk_cur[aidx]  = f2bf(snew);
        sfPk_cur[aidx] = f2bf(gdv);

        // readout = s_new @ O^T : reduce over the 16 lanes sharing row m
        float r0 = snew * o0, r1 = snew * o1;
        #pragma unroll
        for (int msk = 1; msk <= 8; msk <<= 1) {
            r0 += __shfl_xor(r0, msk, 32);
            r1 += __shfl_xor(r1, msk, 32);
        }
        if (l15 == 0) {
            atomicAdd(&readout[m * 2 + 0], r0);
            atomicAdd(&readout[m * 2 + 1], r1);
        }
    }
}

// ---------------------------------------------------------------------------
extern "C" void kernel_launch(void* const* d_in, const int* in_sizes, int n_in,
                              void* d_out, int out_size, void* d_ws, size_t ws_size,
                              hipStream_t stream) {
    (void)in_sizes; (void)n_in; (void)out_size; (void)ws_size;

    const float* x_in  = (const float*)d_in[0];   // [128,64,2]
    const float* Wsyn  = (const float*)d_in[1];   // [1024,1024]
    const float* Wfast = (const float*)d_in[2];   // [1024,1024]
    const float* W_in  = (const float*)d_in[3];   // [2,1024]
    const float* O     = (const float*)d_in[4];   // [2,1024]
    float* out = (float*)d_out;                   // spikes [T,B,N] ++ readout [T,B,2]

    // workspace carve-out (~5.25 MB)
    uint8_t* ws = (uint8_t*)d_ws;
    size_t off = 0;
    __bf16* Wsb = (__bf16*)(ws + off); off += (size_t)NN * NN * 2;      // 2 MB
    __bf16* Wfb = (__bf16*)(ws + off); off += (size_t)NN * NN * 2;      // 2 MB
    float*  sF32[2]; __bf16* sPk[2]; __bf16* sfPk[2];
    sF32[0] = (float*)(ws + off);  off += (size_t)NB * NN * 4;
    sF32[1] = (float*)(ws + off);  off += (size_t)NB * NN * 4;
    sPk[0]  = (__bf16*)(ws + off); off += (size_t)NB * NN * 2;
    sPk[1]  = (__bf16*)(ws + off); off += (size_t)NB * NN * 2;
    sfPk[0] = (__bf16*)(ws + off); off += (size_t)NB * NN * 2;
    sfPk[1] = (__bf16*)(ws + off); off += (size_t)NB * NN * 2;
    float* vpk = (float*)(ws + off); off += (size_t)NB * NN * 4;

    float* readout_base = out + (size_t)NT * NB * NN;

    nlif_pack_weights<<<8192, 256, 0, stream>>>(Wsyn, Wfast, Wsb, Wfb);
    nlif_init<<<256, 256, 0, stream>>>(sF32[0], sPk[0], sfPk[0], vpk, readout_base);

    for (int t = 0; t < NT; ++t) {
        const int pi = t & 1, ci = pi ^ 1;
        nlif_step<<<32, 256, 0, stream>>>(
            x_in + (size_t)t * NB * 2, W_in, O, Wsb, Wfb,
            sF32[pi], sPk[pi], sfPk[pi],
            sF32[ci], sPk[ci], sfPk[ci],
            vpk,
            out + (size_t)t * NB * NN,
            readout_base + (size_t)t * NB * 2);
    }
}
